// EpisodicMemorySSM_69226282877356
// MI455X (gfx1250) — matrix operand order repaired and checked by reading, not compiled, not run on hardware
//
#include <hip/hip_runtime.h>
#include <hip/hip_bf16.h>

// ---------------------------------------------------------------------------
// EpisodicMemorySSM forward for MI455X (gfx1250, wave32, WMMA).
// GEMMs: bf16 x bf16 -> f32 via V_WMMA_F32_16X16X32_BF16, with double-buffered
// GLOBAL_LOAD_ASYNC_TO_LDS_B128 staging of the 64x32 A/B tiles (ASYNCcnt).
// Selective scan: one wave per channel, lane n = state h[e,n] (N=32 lanes).
// ---------------------------------------------------------------------------

typedef __bf16 bf16_t;
typedef __attribute__((ext_vector_type(16))) __bf16 v16bf;
typedef __attribute__((ext_vector_type(8)))  float  v8f;

// 16-byte payload type expected by the async-to-LDS builtins.
typedef int v4i_ls __attribute__((vector_size(4 * sizeof(int))));
typedef __attribute__((address_space(1))) v4i_ls* gptr16_t;
typedef __attribute__((address_space(3))) v4i_ls* lptr16_t;

static constexpr int NBq  = 4;
static constexpr int Dq   = 1024;
static constexpr int EDq  = 2048;
static constexpr int Nq   = 32;    // SSM state size == wave32 lanes
static constexpr int Kcq  = 4;     // conv kernel
static constexpr int DTRq = 64;
static constexpr int Lq   = 1024;
static constexpr int XDBL = DTRq + 2 * Nq;  // 128

#if __has_builtin(__builtin_amdgcn_global_load_async_to_lds_b128) && \
    __has_builtin(__builtin_amdgcn_s_wait_asynccnt)
#define USE_ASYNC_LDS 1
#else
#define USE_ASYNC_LDS 0
#endif

// ---------------- WMMA fragment loaders (contiguous k order, same in A and B) ----------------
// A: 16x32 bf16, row-major source, leading dim lda.
// lane(0..15)=row M, hi=lane>>4: elems 0..15 -> K=hi*16+i.
__device__ __forceinline__ v16bf load_a_frag(const bf16_t* __restrict__ A, int lda) {
  int lane = threadIdx.x & 31;
  const bf16_t* p = A + (size_t)(lane & 15) * lda + ((lane >> 4) << 4);
  v16bf r;
#pragma unroll
  for (int i = 0; i < 16; ++i) r[i] = p[i];
  return r;
}

// B: 32x16 bf16 where B(k,n) = W(n,k), W row-major (N x K), leading dim ldw.
// lane(0..15)=col n, hi selects K half -> contiguous 16 bf16 of row n.
__device__ __forceinline__ v16bf load_b_frag(const bf16_t* __restrict__ W, int ldw) {
  int lane = threadIdx.x & 31;
  const bf16_t* p = W + (size_t)(lane & 15) * ldw + ((lane >> 4) << 4);
  v16bf r;
#pragma unroll
  for (int i = 0; i < 16; ++i) r[i] = p[i];
  return r;
}

// C = A(MxK) @ W(NxK)^T, f32 accumulate. Wave: 16x64 strip (4 WMMA tiles).
// Block: 128 thr = 4 waves stacked in M -> 64x64 block tile. Grid: (N/64, M/64).
// M, N multiples of 64; K multiple of 32 (holds for all call sites).
__global__ __launch_bounds__(128) void k_gemm_bf16(
    const bf16_t* __restrict__ A, const bf16_t* __restrict__ W,
    float* __restrict__ C, int M, int N, int K) {
  int wave = threadIdx.x >> 5;
  int lane = threadIdx.x & 31;
  int n0 = blockIdx.x << 6;
  int m0b = blockIdx.y << 6;
  int m0 = m0b + (wave << 4);
  v8f acc[4] = {};

#if USE_ASYNC_LDS
  // Double-buffered 64x32 bf16 tiles of A and of W (B-operand), 16 KB total.
  __shared__ __align__(16) bf16_t As[2][64][32];
  __shared__ __align__(16) bf16_t Bs[2][64][32];
  int tid = threadIdx.x;
  const int nk = K >> 5;

  auto fill = [&](int buf, int k) {
    // 64x32 bf16 tile = 256 chunks of 16B; 128 threads x 2 chunks each.
#pragma unroll
    for (int c = tid; c < 256; c += 128) {
      int row = c >> 2;
      int col = (c & 3) << 3;
      __builtin_amdgcn_global_load_async_to_lds_b128(
          (gptr16_t)(A + (size_t)(m0b + row) * K + k + col),
          (lptr16_t)&As[buf][row][col], 0, 0);
      __builtin_amdgcn_global_load_async_to_lds_b128(
          (gptr16_t)(W + (size_t)(n0 + row) * K + k + col),
          (lptr16_t)&Bs[buf][row][col], 0, 0);
    }
  };

  fill(0, 0);
  for (int kb = 0; kb < nk; ++kb) {
    int cur = kb & 1;
    __builtin_amdgcn_s_wait_asynccnt(0);  // this thread's fill of `cur` landed
    __syncthreads();                      // all threads' fills landed; all reads
                                          // of buffer cur^1 from step kb-1 done
    if (kb + 1 < nk) fill(cur ^ 1, (kb + 1) << 5);  // overlap with compute
    v16bf a;
    {
      const bf16_t* p = &As[cur][(wave << 4) + (lane & 15)][(lane >> 4) << 4];
#pragma unroll
      for (int i = 0; i < 16; ++i) a[i] = p[i];
    }
#pragma unroll
    for (int t = 0; t < 4; ++t) {
      v16bf b;
      const bf16_t* q = &Bs[cur][(t << 4) + (lane & 15)][(lane >> 4) << 4];
#pragma unroll
      for (int i = 0; i < 16; ++i) b[i] = q[i];
      acc[t] = __builtin_amdgcn_wmma_f32_16x16x32_bf16(
          false, a, false, b, (short)0, acc[t], false, false);
    }
  }
#else
  const bf16_t* Ap = A + (size_t)m0 * K;
  for (int k = 0; k < K; k += 32) {
    if (k + 32 < K) __builtin_prefetch(Ap + (size_t)k + 32, 0, 1);
    v16bf a = load_a_frag(Ap + k, K);
#pragma unroll
    for (int t = 0; t < 4; ++t) {
      v16bf b = load_b_frag(W + (size_t)(n0 + t * 16) * K + k, K);
      acc[t] = __builtin_amdgcn_wmma_f32_16x16x32_bf16(
          false, a, false, b, (short)0, acc[t], false, false);
    }
  }
#endif

  int hi = lane >> 4, nn = lane & 15;
#pragma unroll
  for (int t = 0; t < 4; ++t)
#pragma unroll
    for (int v = 0; v < 8; ++v)
      C[(size_t)(m0 + v + 8 * hi) * N + (n0 + t * 16 + nn)] = acc[t][v];
}

// ---------------- elementwise / conversion kernels ----------------

__global__ void k_cvt_bf16(const float* __restrict__ s, bf16_t* __restrict__ d, int n) {
  int i = blockIdx.x * blockDim.x + threadIdx.x;
  if (i < n) d[i] = (bf16_t)s[i];
}

// strided convert: d[r*cols+c] = s[r*ld+c]
__global__ void k_cvt_bf16_2d(const float* __restrict__ s, bf16_t* __restrict__ d,
                              int rows, int cols, int ld) {
  int i = blockIdx.x * blockDim.x + threadIdx.x;
  if (i >= rows * cols) return;
  int r = i / cols, c = i - r * cols;
  d[i] = (bf16_t)s[(size_t)r * ld + c];
}

__global__ void k_bias_add(float* __restrict__ h, const float* __restrict__ b,
                           int rows, int cols) {
  int i = blockIdx.x * blockDim.x + threadIdx.x;
  if (i < rows * cols) h[i] += b[i % cols];
}

__global__ void k_softplus_bias(float* __restrict__ x, const float* __restrict__ b,
                                int rows, int cols) {
  int i = blockIdx.x * blockDim.x + threadIdx.x;
  if (i >= rows * cols) return;
  float v = x[i] + b[i % cols];
  x[i] = (v > 20.f) ? v : log1pf(__expf(v));
}

__global__ void k_resid_add(float* __restrict__ h, const float* __restrict__ m, int n) {
  int i = blockIdx.x * blockDim.x + threadIdx.x;
  if (i < n) h[i] += m[i];
}

// causal depthwise conv (K=4) + bias + SiLU; u taken from xz[:, 0:ED]
__global__ void k_conv_silu(const float* __restrict__ xz, const float* __restrict__ cw,
                            const float* __restrict__ cb, float* __restrict__ uf,
                            bf16_t* __restrict__ ub, int L, int ED) {
  int i = blockIdx.x * blockDim.x + threadIdx.x;
  if (i >= L * ED) return;
  int e = i % ED, l = i / ED;
  float acc = cb[e];
#pragma unroll
  for (int j = 0; j < Kcq; ++j) {
    int ls = l - (Kcq - 1) + j;
    if (ls >= 0) acc += cw[e * Kcq + j] * xz[(size_t)ls * (2 * ED) + e];
  }
  float v = acc / (1.f + __expf(-acc));
  uf[i] = v;
  ub[i] = (bf16_t)v;
}

// y = (ys + u*Dp[e]) * silu(z);   z = xz[:, ED:2ED]
__global__ void k_ygate(const float* __restrict__ ys, const float* __restrict__ uf,
                        const float* __restrict__ xz, const float* __restrict__ Dp,
                        bf16_t* __restrict__ yb, int L, int ED) {
  int i = blockIdx.x * blockDim.x + threadIdx.x;
  if (i >= L * ED) return;
  int e = i % ED, l = i / ED;
  float z = xz[(size_t)l * (2 * ED) + ED + e];
  float sz = z / (1.f + __expf(-z));
  yb[i] = (bf16_t)((ys[i] + uf[i] * Dp[e]) * sz);
}

// ---------------- layernorm (row = D elems, block per row) ----------------

template <typename OUT>
__global__ __launch_bounds__(256) void k_layernorm(const float* __restrict__ x,
                                                   const float* __restrict__ g,
                                                   const float* __restrict__ b,
                                                   OUT* __restrict__ out, int D) {
  int row = blockIdx.x;
  const float* xr = x + (size_t)row * D;
  __shared__ float sb[2][8];
  float s = 0.f, s2 = 0.f;
  for (int i = threadIdx.x; i < D; i += blockDim.x) {
    float v = xr[i]; s += v; s2 += v * v;
  }
  for (int off = 16; off > 0; off >>= 1) {
    s  += __shfl_xor(s,  off, 32);
    s2 += __shfl_xor(s2, off, 32);
  }
  int wid = threadIdx.x >> 5;
  if ((threadIdx.x & 31) == 0) { sb[0][wid] = s; sb[1][wid] = s2; }
  __syncthreads();
  float ts = 0.f, ts2 = 0.f;
#pragma unroll
  for (int j = 0; j < 8; ++j) { ts += sb[0][j]; ts2 += sb[1][j]; }
  float mean = ts / D;
  float var = ts2 / D - mean * mean;
  float rstd = rsqrtf(var + 1e-5f);
  for (int i = threadIdx.x; i < D; i += blockDim.x)
    out[(size_t)row * D + i] = (OUT)((xr[i] - mean) * rstd * g[i] + b[i]);
}

// ---------------- selective scan: one wave per channel ----------------
// lane n holds h[e,n]; B,C read from xdbl (stride 128, offsets 64/96).
__global__ __launch_bounds__(256) void k_scan(const float* __restrict__ delta,
                                              const float* __restrict__ uf,
                                              const float* __restrict__ xdbl,
                                              const float* __restrict__ A_log,
                                              float* __restrict__ ys, int L, int ED) {
  int gw = (blockIdx.x * blockDim.x + threadIdx.x) >> 5;
  int lane = threadIdx.x & 31;
  if (gw >= ED) return;
  int e = gw;
  float A = -__expf(A_log[(size_t)e * Nq + lane]);
  float h = 0.f;
  for (int l = 0; l < L; ++l) {
    float dlt = delta[(size_t)l * ED + e];
    float uu  = uf[(size_t)l * ED + e];
    float Bv  = xdbl[(size_t)l * XDBL + DTRq + lane];
    float Cv  = xdbl[(size_t)l * XDBL + DTRq + Nq + lane];
    h = __expf(dlt * A) * h + (dlt * uu) * Bv;
    float yv = h * Cv;
#pragma unroll
    for (int off = 16; off > 0; off >>= 1) yv += __shfl_xor(yv, off, 32);
    if (lane == 0) ys[(size_t)l * ED + e] = yv;
  }
}

// ---------------- head: column mean + tiny GEMV ----------------

__global__ void k_colmean(const float* __restrict__ hf, float* __restrict__ cm,
                          int L, int D) {
  int d = blockIdx.x * blockDim.x + threadIdx.x;
  if (d >= D) return;
  float s = 0.f;
  for (int l = 0; l < L; ++l) s += hf[(size_t)l * D + d];
  cm[d] = s / L;
}

__global__ __launch_bounds__(256) void k_state(const float* __restrict__ cm,
                                               const float* __restrict__ spw,
                                               const float* __restrict__ spb,
                                               float* __restrict__ st, int D) {
  int n = blockIdx.x;
  __shared__ float sb[8];
  float s = 0.f;
  for (int d = threadIdx.x; d < D; d += blockDim.x) s += cm[d] * spw[(size_t)n * D + d];
  for (int off = 16; off > 0; off >>= 1) s += __shfl_xor(s, off, 32);
  if ((threadIdx.x & 31) == 0) sb[threadIdx.x >> 5] = s;
  __syncthreads();
  if (threadIdx.x == 0) {
    float t = 0.f;
#pragma unroll
    for (int j = 0; j < 8; ++j) t += sb[j];
    st[n] = t + spb[n];
  }
}

// ---------------------------------------------------------------------------

static inline dim3 gemm_grid(int M, int N) { return dim3(N / 64, M / 64); }
static inline int  ceil_div(int a, int b)  { return (a + b - 1) / b; }

extern "C" void kernel_launch(void* const* d_in, const int* in_sizes, int n_in,
                              void* d_out, int out_size, void* d_ws, size_t ws_size,
                              hipStream_t stream) {
  const float* x        = (const float*)d_in[0];
  const float* Wi       = (const float*)d_in[1];
  const float* bi       = (const float*)d_in[2];
  const float* ln_g     = (const float*)d_in[3];
  const float* ln_b     = (const float*)d_in[4];
  const float* in_w     = (const float*)d_in[5];
  const float* conv_w   = (const float*)d_in[6];
  const float* conv_b   = (const float*)d_in[7];
  const float* xproj_w  = (const float*)d_in[8];
  const float* dtproj_w = (const float*)d_in[9];
  const float* dtproj_b = (const float*)d_in[10];
  const float* A_log    = (const float*)d_in[11];
  const float* Dp       = (const float*)d_in[12];
  const float* out_w    = (const float*)d_in[13];
  const float* fn_g     = (const float*)d_in[14];
  const float* fn_b     = (const float*)d_in[15];
  const float* sp_w     = (const float*)d_in[16];
  const float* sp_b     = (const float*)d_in[17];
  float* out = (float*)d_out;

  // ---- workspace carve-up ----
  char* wp = (char*)d_ws;
  auto alloc = [&](size_t bytes) {
    char* p = wp;
    wp += (bytes + 255) & ~(size_t)255;
    return p;
  };
  float*  h     = (float*)alloc((size_t)Lq * Dq * 4);           // residual stream
  bf16_t* wbf   = (bf16_t*)alloc((size_t)2 * EDq * Dq * 2);     // bf16 weight staging
  bf16_t* actbf = (bf16_t*)alloc((size_t)Lq * Dq * 2);          // x / normed (bf16)
  bf16_t* ubf   = (bf16_t*)alloc((size_t)Lq * EDq * 2);
  bf16_t* ybf   = (bf16_t*)alloc((size_t)Lq * EDq * 2);
  bf16_t* dtbf  = (bf16_t*)alloc((size_t)Lq * DTRq * 2);
  float*  xz    = (float*)alloc((size_t)Lq * 2 * EDq * 4);
  float*  uf    = (float*)alloc((size_t)Lq * EDq * 4);
  float*  xdbl  = (float*)alloc((size_t)Lq * XDBL * 4);
  float*  delta = (float*)alloc((size_t)Lq * EDq * 4);
  float*  ys    = (float*)alloc((size_t)Lq * EDq * 4);
  float*  mout  = (float*)alloc((size_t)Lq * Dq * 4);
  float*  cmean = (float*)alloc((size_t)Dq * 4);

  const int T = 256;
  auto gr = [&](int n) { return ceil_div(n, T); };

  // ---- h = x @ Wi^T + bi ----
  k_cvt_bf16<<<gr(Lq * Dq), T, 0, stream>>>(x, actbf, Lq * Dq);
  k_cvt_bf16<<<gr(Dq * Dq), T, 0, stream>>>(Wi, wbf, Dq * Dq);
  k_gemm_bf16<<<gemm_grid(Lq, Dq), 128, 0, stream>>>(actbf, wbf, h, Lq, Dq, Dq);
  k_bias_add<<<gr(Lq * Dq), T, 0, stream>>>(h, bi, Lq, Dq);

  for (int blk = 0; blk < NBq; ++blk) {
    // normed (bf16) = LN(h)
    k_layernorm<bf16_t><<<Lq, 256, 0, stream>>>(h, ln_g + blk * Dq, ln_b + blk * Dq,
                                                actbf, Dq);
    // xz = normed @ in_w^T   (L x 4096)
    k_cvt_bf16<<<gr(2 * EDq * Dq), T, 0, stream>>>(in_w + (size_t)blk * 2 * EDq * Dq,
                                                   wbf, 2 * EDq * Dq);
    k_gemm_bf16<<<gemm_grid(Lq, 2 * EDq), 128, 0, stream>>>(actbf, wbf, xz,
                                                            Lq, 2 * EDq, Dq);
    // u = silu(causal_conv(xz[:, :ED]) + cb)
    k_conv_silu<<<gr(Lq * EDq), T, 0, stream>>>(xz, conv_w + (size_t)blk * EDq * Kcq,
                                                conv_b + blk * EDq, uf, ubf, Lq, EDq);
    // x_dbl = u @ xproj^T   (L x 128)
    k_cvt_bf16<<<gr(XDBL * EDq), T, 0, stream>>>(xproj_w + (size_t)blk * XDBL * EDq,
                                                 wbf, XDBL * EDq);
    k_gemm_bf16<<<gemm_grid(Lq, XDBL), 128, 0, stream>>>(ubf, wbf, xdbl, Lq, XDBL, EDq);
    // delta = softplus(dt @ dtproj^T + b)
    k_cvt_bf16_2d<<<gr(Lq * DTRq), T, 0, stream>>>(xdbl, dtbf, Lq, DTRq, XDBL);
    k_cvt_bf16<<<gr(EDq * DTRq), T, 0, stream>>>(dtproj_w + (size_t)blk * EDq * DTRq,
                                                 wbf, EDq * DTRq);
    k_gemm_bf16<<<gemm_grid(Lq, EDq), 128, 0, stream>>>(dtbf, wbf, delta, Lq, EDq, DTRq);
    k_softplus_bias<<<gr(Lq * EDq), T, 0, stream>>>(delta, dtproj_b + blk * EDq, Lq, EDq);
    // selective scan (2048 waves, one per channel)
    k_scan<<<ceil_div(EDq * 32, 256), 256, 0, stream>>>(delta, uf, xdbl,
                                                        A_log + (size_t)blk * EDq * Nq,
                                                        ys, Lq, EDq);
    // y = (ys + u*Dp) * silu(z)  (bf16)
    k_ygate<<<gr(Lq * EDq), T, 0, stream>>>(ys, uf, xz, Dp + blk * EDq, ybf, Lq, EDq);
    // mout = y @ out_w^T ; h += mout
    k_cvt_bf16<<<gr(Dq * EDq), T, 0, stream>>>(out_w + (size_t)blk * Dq * EDq,
                                               wbf, Dq * EDq);
    k_gemm_bf16<<<gemm_grid(Lq, Dq), 128, 0, stream>>>(ybf, wbf, mout, Lq, Dq, EDq);
    k_resid_add<<<gr(Lq * Dq), T, 0, stream>>>(h, mout, Lq * Dq);
  }

  // ---- final LN -> d_out, then state head ----
  k_layernorm<float><<<Lq, 256, 0, stream>>>(h, fn_g, fn_b, out, Dq);
  k_colmean<<<gr(Dq), T, 0, stream>>>(out, cmean, Lq, Dq);
  k_state<<<Nq, 256, 0, stream>>>(cmean, sp_w, sp_b, out + (size_t)Lq * Dq, Dq);
}
